// TextBinOp_11261404250810
// MI455X (gfx1250) — compile-verified, matrix operand order checked by reading, not run on hardware
//
#include <hip/hip_runtime.h>

#define EMB 64
#define LL 48
#define PADW 56          // LL + 2*4 halo, zero-padded activation layout
#define NB_BATCH 4
#define BN_EPS 1e-3f

typedef _Float16 v8h  __attribute__((ext_vector_type(8)));
typedef _Float16 v16h __attribute__((ext_vector_type(16)));
typedef float    v8f  __attribute__((ext_vector_type(8)));

// ---------------------------------------------------------------------------
// Zero-fill (16B granular) for the padded activation buffers.
// ---------------------------------------------------------------------------
__global__ void zero_kernel(uint4* __restrict__ p, long n16) {
  long idx = (long)blockIdx.x * blockDim.x + threadIdx.x;
  if (idx < n16) { uint4 z = {0u, 0u, 0u, 0u}; p[idx] = z; }
}

// ---------------------------------------------------------------------------
// Fold BN (inference) into per-channel scale/shift:
//   scale = gamma * rsqrt(var+eps); shift = beta - mean*scale
// ---------------------------------------------------------------------------
__global__ void bn_fold_kernel(const float* __restrict__ g, const float* __restrict__ be,
                               const float* __restrict__ mu, const float* __restrict__ va,
                               float* __restrict__ scale, float* __restrict__ shift, int cin) {
  int c = blockIdx.x * blockDim.x + threadIdx.x;
  if (c < cin) {
    float s = g[c] * rsqrtf(va[c] + BN_EPS);
    scale[c] = s;
    shift[c] = be[c] - mu[c] * s;
  }
}

// Identity scale/shift for the last conv's epilogue (no following BN).
__global__ void id_kernel(float* __restrict__ s, float* __restrict__ t) {
  int c = threadIdx.x;
  s[c] = 1.0f;
  t[c] = 0.0f;
}

// ---------------------------------------------------------------------------
// Weight convert + transpose: W[81, cin, cout] f32 -> Wt[81, cout, cin] f16
// (a WMMA B-fragment lane then reads two contiguous 16B chunks)
// ---------------------------------------------------------------------------
__global__ void wcvt_kernel(const float* __restrict__ W, _Float16* __restrict__ Wt,
                            int cin, int cout) {
  int idx = blockIdx.x * blockDim.x + threadIdx.x;
  int total = 81 * cin * cout;
  if (idx >= total) return;
  int ci = idx % cin;
  int t  = idx / cin;
  int co = t % cout;
  int kk = t / cout;
  Wt[idx] = (_Float16)W[((long)kk * cin + ci) * cout + co];
}

// ---------------------------------------------------------------------------
// Pairwise grid with layer-0 BN fused, stored f16 into padded layout:
//   m[b, i+4, j+4, c] = BN0( (c<EMB)? x[b,i,c] : y[b,j,c-EMB] )
// ---------------------------------------------------------------------------
__global__ void grid_kernel(const float* __restrict__ x, const float* __restrict__ y,
                            const float* __restrict__ s0, const float* __restrict__ t0,
                            _Float16* __restrict__ m) {
  int idx = blockIdx.x * blockDim.x + threadIdx.x;
  const int C = 2 * EMB;
  int total = NB_BATCH * LL * LL * C;
  if (idx >= total) return;
  int c = idx % C;
  int t = idx / C;
  int j = t % LL; t /= LL;
  int i = t % LL;
  int b = t / LL;
  float v = (c < EMB) ? x[(b * LL + i) * EMB + c] : y[(b * LL + j) * EMB + (c - EMB)];
  m[(((long)b * PADW + (i + 4)) * PADW + (j + 4)) * C + c] = (_Float16)(v * s0[c] + t0[c]);
}

// ---------------------------------------------------------------------------
// Implicit-GEMM 9x9 SAME conv via WMMA f16, fully fused pipeline:
//   in : [NB,56,56,cin]  f16, zero-padded borders, BN already applied
//   Wt : [81, cout, cin] f16 transposed weights
//   out: [NB,56,56,cout] f16; epilogue = bias + ReLU + *next* layer's BN
//
// Block = (b, NA-row tile, 16-col tile); 4 waves; wave owns 16*NBT couts.
// Wave holds NA*NBT accumulators: B fragment (weights, L2) reused NA times
// (weight AI = 16*NA F/B), A fragment (LDS) reused NBT times.
// The (NA+8)x24xcin f16 window is staged to LDS ONCE per block with
// global_load_async_to_lds_b128 (ASYNCcnt path, no VGPR round-trip), then
// 9x9x(cin/32) K-loop of WMMA_F32_16X16X32_F16 runs barrier-free.
// ---------------------------------------------------------------------------
#define LDS_PAD 8
template <int NA, int NBT>
__global__ __launch_bounds__(128) void conv9x9_wmma_kernel(
    const _Float16* __restrict__ in,
    const _Float16* __restrict__ Wt,
    const float*    __restrict__ bias,
    const float*    __restrict__ sNext,
    const float*    __restrict__ tNext,
    _Float16*       __restrict__ out,
    int cin, int cout)
{
  extern __shared__ __align__(16) _Float16 sA[];  // (NA+8) rows * 24 cols * (cin+PAD)
  const int ldsStride = cin + LDS_PAD;
  const int rowStride = 24 * ldsStride;

  const int tid   = threadIdx.x;
  const int lane  = tid & 31;
  const int wave  = tid >> 5;
  const int n0    = wave * 16 * NBT;
  const int nlo   = lane & 15;   // N for B/C fragments, M for A fragment
  const int khalf = lane >> 4;   // hi/lo half of the wave

  int bid = blockIdx.x;
  const int jt = bid % (LL / 16); bid /= (LL / 16);
  const int it = bid % (LL / NA);
  const int b  = bid / (LL / NA);
  const int j0 = jt * 16;
  const int i0 = it * NA;

  // ---- async-stage the (NA+8) x 24 x cin f16 window into LDS (pure copy) ----
  {
    const int chunksPerPix = cin >> 3;                 // 16B chunks per pixel
    const int nChunks = (NA + 8) * 24 * chunksPerPix;  // divisible by 128
    const unsigned ldsBase = (unsigned)(size_t)(&sA[0]);
    for (int tr = tid; tr < nChunks; tr += 128) {
      const int pix = tr / chunksPerPix;
      const int c   = (tr - pix * chunksPerPix) << 3;
      const int wr  = pix / 24;
      const int col = pix - wr * 24;
      const unsigned loff = ldsBase + (unsigned)((wr * rowStride + col * ldsStride + c) * 2);
      const _Float16* g = in + (((long)b * PADW + (i0 + wr)) * PADW + (j0 + col)) * cin + c;
      const unsigned long long ga = (unsigned long long)(size_t)g;
      asm volatile("global_load_async_to_lds_b128 %0, %1, off"
                   :: "v"(loff), "v"(ga) : "memory");
    }
    asm volatile("s_wait_asynccnt 0x0" ::: "memory");
  }
  __syncthreads();

  v8f acc[NA][NBT];
#pragma unroll
  for (int d = 0; d < NA; ++d)
#pragma unroll
    for (int e = 0; e < NBT; ++e) acc[d][e] = (v8f){};

  for (int kh = 0; kh < 9; ++kh) {
    for (int kw = 0; kw < 9; ++kw) {
      // A source: window row d+kh, local col nlo+kw
      const _Float16* arow0 = &sA[kh * rowStride + (nlo + kw) * ldsStride + khalf * 8];
      // B source: Wt[(kh*9+kw), n0 + e*16 + nlo, :]
      const _Float16* brow0 = &Wt[((long)(kh * 9 + kw) * cout + (n0 + nlo)) * cin + khalf * 8];
      for (int ck = 0; ck < cin; ck += 32) {
        v16h Af[NA], Bf[NBT];
#pragma unroll
        for (int d = 0; d < NA; ++d) {
          const _Float16* a = arow0 + d * rowStride + ck;
          v8h a0 = *(const v8h*)(a);
          v8h a1 = *(const v8h*)(a + 16);
#pragma unroll
          for (int q = 0; q < 8; ++q) { Af[d][q] = a0[q]; Af[d][q + 8] = a1[q]; }
        }
#pragma unroll
        for (int e = 0; e < NBT; ++e) {
          const _Float16* bp = brow0 + (long)e * 16 * cin + ck;
          v8h b0 = *(const v8h*)(bp);
          v8h b1 = *(const v8h*)(bp + 16);
#pragma unroll
          for (int q = 0; q < 8; ++q) { Bf[e][q] = b0[q]; Bf[e][q + 8] = b1[q]; }
        }
#pragma unroll
        for (int e = 0; e < NBT; ++e)
#pragma unroll
          for (int d = 0; d < NA; ++d)
            acc[d][e] = __builtin_amdgcn_wmma_f32_16x16x32_f16(
                false, Af[d], false, Bf[e], (short)0, acc[d][e], false, false);
      }
    }
  }

  // ---- epilogue: bias + ReLU + next layer's BN, store f16 into padded out ----
  // C/D layout: lane (nlo, khalf), vgpr rr -> (M = khalf*8+rr, N = nlo)
#pragma unroll
  for (int e = 0; e < NBT; ++e) {
    const int ch = n0 + e * 16 + nlo;
    const float bn = bias[ch];
    const float s  = sNext[ch];
    const float t  = tNext[ch];
#pragma unroll
    for (int d = 0; d < NA; ++d) {
#pragma unroll
      for (int rr = 0; rr < 8; ++rr) {
        const int m = khalf * 8 + rr;
        float v = acc[d][e][rr] + bn;
        v = v > 0.0f ? v : 0.0f;
        out[(((long)b * PADW + (i0 + d + 4)) * PADW + (j0 + m + 4)) * cout + ch] =
            (_Float16)(v * s + t);
      }
    }
  }
}

// ---------------------------------------------------------------------------
// out[b,j,c] = max_i act[b, i+4, j+4, c]   act: padded f16 [NB,56,56,EMB]
// ---------------------------------------------------------------------------
__global__ void maxred_kernel(const _Float16* __restrict__ act, float* __restrict__ out) {
  int idx = blockIdx.x * blockDim.x + threadIdx.x;
  int total = NB_BATCH * LL * EMB;
  if (idx >= total) return;
  int c = idx % EMB;
  int t = idx / EMB;
  int j = t % LL;
  int b = t / LL;
  float m = -3.402823466e38f;
  for (int i = 0; i < LL; ++i)
    m = fmaxf(m, (float)act[(((long)b * PADW + (i + 4)) * PADW + (j + 4)) * EMB + c]);
  out[idx] = m;
}

// ---------------------------------------------------------------------------
extern "C" void kernel_launch(void* const* d_in, const int* in_sizes, int n_in,
                              void* d_out, int out_size, void* d_ws, size_t ws_size,
                              hipStream_t stream) {
  static const int CIN[6]  = {128, 128, 128, 128, 64, 64};
  static const int COUT[6] = {128, 128, 128,  64, 64, 64};
  const int NA = 4;  // rows per block / accumulators sharing one B fragment

  const float *x = nullptr, *y = nullptr;
  const float *gamma[6], *beta[6], *mean[6], *var[6], *W[6], *bias[6];

  if (in_sizes[0] == NB_BATCH * LL * EMB) {
    // insertion order: x, y, then per layer gamma,beta,mean,var,W,b
    x = (const float*)d_in[0];
    y = (const float*)d_in[1];
    int k = 2;
    for (int l = 0; l < 6; ++l) {
      gamma[l] = (const float*)d_in[k++];
      beta[l]  = (const float*)d_in[k++];
      mean[l]  = (const float*)d_in[k++];
      var[l]   = (const float*)d_in[k++];
      W[l]     = (const float*)d_in[k++];
      bias[l]  = (const float*)d_in[k++];
    }
  } else {
    // sorted pytree order: per layer W,b,beta,gamma,mean,var; then x, y
    int k = 0;
    for (int l = 0; l < 6; ++l) {
      W[l]     = (const float*)d_in[k++];
      bias[l]  = (const float*)d_in[k++];
      beta[l]  = (const float*)d_in[k++];
      gamma[l] = (const float*)d_in[k++];
      mean[l]  = (const float*)d_in[k++];
      var[l]   = (const float*)d_in[k++];
    }
    x = (const float*)d_in[k++];
    y = (const float*)d_in[k++];
  }

  // ---- workspace carve (all chunks 16B-aligned) ----
  char* p = (char*)d_ws;
  const size_t pad128Bytes = (size_t)NB_BATCH * PADW * PADW * 128 * sizeof(_Float16);
  const size_t pad64Bytes  = (size_t)NB_BATCH * PADW * PADW *  64 * sizeof(_Float16);
  _Float16* actZeroBase = (_Float16*)p;
  _Float16* p128A = (_Float16*)p; p += pad128Bytes;
  _Float16* p128B = (_Float16*)p; p += pad128Bytes;
  _Float16* p64A  = (_Float16*)p; p += pad64Bytes;
  _Float16* p64B  = (_Float16*)p; p += pad64Bytes;
  const size_t actTotalBytes = 2 * pad128Bytes + 2 * pad64Bytes;
  _Float16* Wt[6];
  for (int l = 0; l < 6; ++l) {
    Wt[l] = (_Float16*)p;
    p += (size_t)81 * CIN[l] * COUT[l] * sizeof(_Float16);
  }
  float *scaleP[6], *shiftP[6];
  for (int l = 0; l < 6; ++l) { scaleP[l] = (float*)p; p += 128 * sizeof(float); }
  for (int l = 0; l < 6; ++l) { shiftP[l] = (float*)p; p += 128 * sizeof(float); }
  float* idScale = (float*)p; p += 128 * sizeof(float);
  float* idShift = (float*)p; p += 128 * sizeof(float);

  // ---- zero padded activation buffers (borders must be 0 every launch) ----
  {
    long n16 = (long)(actTotalBytes / 16);
    zero_kernel<<<(int)((n16 + 255) / 256), 256, 0, stream>>>((uint4*)actZeroBase, n16);
  }

  // ---- parameter prep ----
  for (int l = 0; l < 6; ++l) {
    bn_fold_kernel<<<1, CIN[l], 0, stream>>>(gamma[l], beta[l], mean[l], var[l],
                                             scaleP[l], shiftP[l], CIN[l]);
    int total = 81 * CIN[l] * COUT[l];
    wcvt_kernel<<<(total + 255) / 256, 256, 0, stream>>>(W[l], Wt[l], CIN[l], COUT[l]);
  }
  id_kernel<<<1, 128, 0, stream>>>(idScale, idShift);

  // ---- pairwise grid with layer-0 BN fused ----
  {
    int total = NB_BATCH * LL * LL * 2 * EMB;
    grid_kernel<<<(total + 255) / 256, 256, 0, stream>>>(x, y, scaleP[0], shiftP[0], p128A);
  }

  // ---- 6 conv blocks; tensors: p128A -> p128B -> p128A -> p128B -> p64A -> p64B -> p64A
  _Float16* tin[6]  = {p128A, p128B, p128A, p128B, p64A, p64B};
  _Float16* tout[6] = {p128B, p128A, p128B, p64A,  p64B, p64A};
  for (int l = 0; l < 6; ++l) {
    const int blocks = NB_BATCH * (LL / NA) * (LL / 16);  // 144
    const size_t sh  = (size_t)(NA + 8) * 24 * (CIN[l] + LDS_PAD) * sizeof(_Float16);
    const float* sN = (l < 5) ? scaleP[l + 1] : idScale;
    const float* tN = (l < 5) ? shiftP[l + 1] : idShift;
    if (COUT[l] == 128) {
      conv9x9_wmma_kernel<4, 2><<<blocks, 128, sh, stream>>>(
          tin[l], Wt[l], bias[l], sN, tN, tout[l], CIN[l], COUT[l]);
    } else {
      conv9x9_wmma_kernel<4, 1><<<blocks, 128, sh, stream>>>(
          tin[l], Wt[l], bias[l], sN, tN, tout[l], CIN[l], COUT[l]);
    }
  }

  // ---- max over i ----
  {
    int total = NB_BATCH * LL * EMB;
    maxred_kernel<<<(total + 255) / 256, 256, 0, stream>>>(p64A, (float*)d_out);
  }
}
